// CustomOffsetCriterion_6210522710056
// MI455X (gfx1250) — matrix operand order repaired and checked by reading, not compiled, not run on hardware
//
#include <hip/hip_runtime.h>

typedef __attribute__((ext_vector_type(16))) _Float16 v16h;
typedef __attribute__((ext_vector_type(8)))  float    v8f;
typedef __attribute__((ext_vector_type(8)))  unsigned v8u;

#define Hh    256
#define Ww    256
#define HW    65536
#define Cc    16
#define CO    16
#define KK    9
#define FEAT  144          // C * k*k
#define FPAD  160          // padded K (multiple of 32)
#define ROWU  81           // u32 row stride (162 f16) -> odd => bank-conflict-free
#define WAVES 8

__global__ void zero_out_kernel(float* out) { out[0] = 0.f; }

__global__ __launch_bounds__(256) void deform_conv_loss_kernel(
    const float* __restrict__ offsets, const float* __restrict__ inp,
    const float* __restrict__ ker, const float* __restrict__ tgt,
    float* __restrict__ out)
{
  __shared__ unsigned sB[16 * ROWU];           // ker^T staged as f16 [n][f]
  __shared__ unsigned sA[WAVES * 16 * ROWU];   // per-wave sampled A [16][f] f16
  __shared__ float    red[256];

  const int tid  = threadIdx.x;
  const int lane = tid & 31;
  const int wave = tid >> 5;

  // ---- stage kernel weights (block-cooperative), zero-pad K to 160 ----
  _Float16* sBh = (_Float16*)sB;
  for (int t = tid; t < 16 * FPAD; t += 256) {
    int n = t / FPAD, f = t % FPAD;
    float v = (f < FEAT) ? ker[n * FEAT + f] : 0.f;   // ker[Co][C][3][3] -> [n][c*9+kk]
    sBh[n * (2 * ROWU) + f] = (_Float16)v;
  }

  // ---- per-wave bilinear sampling into LDS A tile ----
  _Float16* sAh = (_Float16*)(sA + wave * 16 * ROWU);
  const int tile = blockIdx.x * WAVES + wave;
  const int g0   = tile * 16;                  // first global pixel of this tile

  for (int ii = 0; ii < 5; ++ii) {
    int pi = ii * 32 + lane;                   // (pixel,tap) pair id, 144 total
    if (pi < FEAT) {
      int p  = pi & 15;                        // pixel in tile
      int kk = pi >> 4;                        // tap 0..8
      int g  = g0 + p;
      int b  = g >> 16;                        // H*W == 65536
      int hw = g & (HW - 1);
      int h  = hw >> 8;
      int w  = hw & (Ww - 1);
      float dy = offsets[((b * 18 + 2 * kk)     << 16) | hw];
      float dx = offsets[((b * 18 + 2 * kk + 1) << 16) | hw];
      int ky = kk / 3, kx = kk - ky * 3;
      float fy = dy + (float)(h - 1 + ky);
      float fx = dx + (float)(w - 1 + kx);
      float y0f = floorf(fy), x0f = floorf(fx);
      float wy = fy - y0f, wx = fx - x0f;
      int iy0 = (int)y0f, ix0 = (int)x0f;
      int iy1 = iy0 + 1,  ix1 = ix0 + 1;
      bool vy0 = (iy0 >= 0) && (iy0 < Hh);
      bool vy1 = (iy1 >= 0) && (iy1 < Hh);
      bool vx0 = (ix0 >= 0) && (ix0 < Ww);
      bool vx1 = (ix1 >= 0) && (ix1 < Ww);
      bool c00 = vy0 && vx0, c01 = vy0 && vx1, c10 = vy1 && vx0, c11 = vy1 && vx1;
      int i00 = iy0 * Ww + ix0;
      int i01 = i00 + 1;
      int i10 = i00 + Ww;
      int i11 = i10 + 1;
      float w00 = (1.f - wy) * (1.f - wx);
      float w01 = (1.f - wy) * wx;
      float w10 = wy * (1.f - wx);
      float w11 = wy * wx;
      const float* cb = inp + (size_t)b * Cc * HW;
      #pragma unroll 4
      for (int c = 0; c < Cc; ++c) {
        const float* pl = cb + c * HW;          // L2-resident gathers
        float s = 0.f;
        if (c00) s += pl[i00] * w00;
        if (c01) s += pl[i01] * w01;
        if (c10) s += pl[i10] * w10;
        if (c11) s += pl[i11] * w11;
        sAh[p * (2 * ROWU) + c * KK + kk] = (_Float16)s;
      }
    }
  }
  // zero-pad A columns 144..159
  for (int t = lane; t < 16 * (FPAD - FEAT); t += 32) {
    int p = t >> 4, col = FEAT + (t & 15);
    sAh[p * (2 * ROWU) + col] = (_Float16)0.f;
  }

  __syncthreads();

  // ---- WMMA: A[16x160]f16 x B[160x16]f16 -> C[16x16]f32, 5 K-chunks ----
  const unsigned* Au = sA + wave * 16 * ROWU;
  const unsigned* Bu = sB;
  const int half = lane >> 4;    // K-half selector per ISA layout
  const int m    = lane & 15;    // row (A) / col (B) within fragment
  v8f acc = {};
  for (int kb = 0; kb < 5; ++kb) {
    v8u au, bu;
    #pragma unroll
    for (int v = 0; v < 8; ++v) {
      // 16-bit A 16x32 layout: pair base K = (v&3)*2 + half*8 + (v>=4)*16
      int Ka = (v & 3) * 2 + (half ? 8 : 0) + ((v >= 4) ? 16 : 0);
      // B 32x16 layout: same K order as A: pair base K = (v&3)*2 + half*8 + (v>=4)*16
      int Kb = (v & 3) * 2 + (half ? 8 : 0) + ((v >= 4) ? 16 : 0);
      au[v] = Au[m * ROWU + ((kb * 32 + Ka) >> 1)];
      bu[v] = Bu[m * ROWU + ((kb * 32 + Kb) >> 1)];
    }
    v16h a   = __builtin_bit_cast(v16h, au);
    v16h bfr = __builtin_bit_cast(v16h, bu);
    acc = __builtin_amdgcn_wmma_f32_16x16x32_f16(false, a, false, bfr,
                                                 (short)0, acc, false, false);
  }

  // ---- squared error vs target; C layout: VGPR r -> M = r + half*8, N = m ----
  float ls = 0.f;
  #pragma unroll
  for (int r = 0; r < 8; ++r) {
    int p  = r + (half ? 8 : 0);
    int g  = g0 + p;
    int b  = g >> 16;
    int hw = g & (HW - 1);
    float t = tgt[(size_t)(b * CO + m) * HW + hw];
    float d = acc[r] - t;
    ls += d * d;
  }

  red[tid] = ls;
  __syncthreads();
  for (int s = 128; s > 0; s >>= 1) {
    if (tid < s) red[tid] += red[tid + s];
    __syncthreads();
  }
  if (tid == 0) atomicAdd(out, red[0] * (1.f / 4194304.f));  // mean over B*Co*H*W
}

extern "C" void kernel_launch(void* const* d_in, const int* in_sizes, int n_in,
                              void* d_out, int out_size, void* d_ws, size_t ws_size,
                              hipStream_t stream) {
  const float* offsets = (const float*)d_in[0];
  const float* inp     = (const float*)d_in[1];
  const float* ker     = (const float*)d_in[2];
  const float* tgt     = (const float*)d_in[3];
  float* out = (float*)d_out;

  zero_out_kernel<<<1, 1, 0, stream>>>(out);
  // 262144 pixels / 16 per wave-tile = 16384 tiles; 8 waves per block
  deform_conv_loss_kernel<<<16384 / WAVES, 256, 0, stream>>>(offsets, inp, ker, tgt, out);
}
